// PiT_25915832664784
// MI455X (gfx1250) — matrix operand order repaired and checked by reading, not compiled, not run on hardware
//
#include <hip/hip_runtime.h>
#include <math.h>

typedef __attribute__((ext_vector_type(16))) _Float16 v16h;
typedef __attribute__((ext_vector_type(8)))  _Float16 v8h;
typedef __attribute__((ext_vector_type(8)))  float    v8f;
typedef __attribute__((ext_vector_type(4)))  float    v4f;

#define NTOK   2048
#define BB     2
#define NH     8
#define VDIM   16
#define HIDD   128
#define MROWS  4096        // B*N
#define LDS_STR 136        // padded A/W row stride (halves)
#define T_STR   72         // padded transpose-tile stride (halves)

__device__ __forceinline__ float gelu_f(float x) {
    return 0.5f * x * (1.0f + erff(x * 0.7071067811865475f));
}

__device__ __forceinline__ v16h cat8(v8h lo, v8h hi) {
    return __builtin_shufflevector(lo, hi, 0,1,2,3,4,5,6,7,8,9,10,11,12,13,14,15);
}

// SWAPX16 within wave32 via ds_swizzle (xor=0x10, and=0x1f): imm = 0x401F
__device__ __forceinline__ float swapx16_f(float x) {
    return __int_as_float(__builtin_amdgcn_ds_swizzle(__float_as_int(x), 0x401F));
}

// ---------------------------------------------------------------------------
// Exact per-row order statistics of m_dist rows (ranks 102 & 103 of 2048),
// blended -> threshold. Binary search on float bit patterns (values in [0,1)).
// ---------------------------------------------------------------------------
__global__ __launch_bounds__(256)
void percentile_kernel(const float* __restrict__ m_dist, float* __restrict__ thr,
                       float frac) {
    __shared__ int red[8];
    const int row = blockIdx.x;                 // 0..4095
    const int tid = threadIdx.x;
    const float* p = m_dist + (size_t)row * NTOK;
    unsigned vals[8];
#pragma unroll
    for (int i = 0; i < 8; ++i) vals[i] = __float_as_uint(p[tid + i * 256]);

    unsigned kth[2];
#pragma unroll 1
    for (int s = 0; s < 2; ++s) {
        const int k = 102 + s;                  // 0-based rank
        unsigned lo = 0u, hi = 0x3F800001u;
#pragma unroll 1
        for (int it = 0; it < 32; ++it) {
            unsigned mid = (lo + hi) >> 1;
            int cnt = 0;
#pragma unroll
            for (int i = 0; i < 8; ++i) cnt += (vals[i] <= mid);
            for (int off = 16; off; off >>= 1) cnt += __shfl_xor(cnt, off, 32);
            if ((tid & 31) == 0) red[tid >> 5] = cnt;
            __syncthreads();
            int total = 0;
#pragma unroll
            for (int w = 0; w < 8; ++w) total += red[w];
            if (total >= k + 1) hi = mid; else lo = mid + 1;
            __syncthreads();
        }
        kth[s] = lo;
    }
    if (tid == 0) {
        float a = __uint_as_float(kth[0]);
        float b = __uint_as_float(kth[1]);
        thr[row] = a * (1.0f - frac) + b * frac;
    }
}

// ---------------------------------------------------------------------------
// Encoder: gelu(inputs(4096x3) @ en_W(3x128) + en_b)
// ---------------------------------------------------------------------------
__global__ __launch_bounds__(256)
void encoder_kernel(const float* __restrict__ x, const float* __restrict__ W,
                    const float* __restrict__ b, float* __restrict__ out) {
    int id = blockIdx.x * 256 + threadIdx.x;
    if (id >= MROWS * HIDD) return;
    int r = id >> 7, c = id & 127;
    float s = b[c] + x[r*3+0]*W[c] + x[r*3+1]*W[HIDD + c] + x[r*3+2]*W[2*HIDD + c];
    out[id] = gelu_f(s);
}

// ---------------------------------------------------------------------------
// (4096x128)@(128x128) WMMA GEMM, f16 operands / f32 accumulate.
// All behavior flags are compile-time -> branch-free epilogues.
//   MODE 0 : out32 = [gelu]( A@W + bias [+ residual] )
//   MODE 1 : f16 transposed-V output via LDS transpose + coalesced b128 stores
//   WLAYOUT 0: W[k*128+c]   1: head-major w[h][k][vd] with c = h*16+vd
// Block: 256 threads = 8 waves. Tile: 64 rows x 128 cols. K = 128.
// ---------------------------------------------------------------------------
template<int MODE, int WLAYOUT, int HAS_RES, int DO_GELU>
__global__ __launch_bounds__(256)
void gemm128_t(const float* __restrict__ A, const float* __restrict__ W,
               const float* __restrict__ bias, const float* __restrict__ residual,
               float* __restrict__ out32, _Float16* __restrict__ out16) {
    __shared__ __align__(16) _Float16 As[64 * LDS_STR];
    __shared__ __align__(16) _Float16 Ws[128 * LDS_STR];   // MODE 1: reused as T[128][T_STR]

    const int tid = threadIdx.x;
    const int m0  = blockIdx.x * 64;

    // Stage A (f32 -> f16), 64x128
    for (int i = tid; i < 64 * 128; i += 256) {
        int r = i >> 7, c = i & 127;
        As[r * LDS_STR + c] = (_Float16)A[(size_t)(m0 + r) * HIDD + c];
    }
    // Stage W transposed: Ws[c][k] = W[k][c]
    for (int i = tid; i < 128 * 128; i += 256) {
        int c = i >> 7, k = i & 127;
        float w = WLAYOUT ? W[(size_t)(c >> 4) * 2048 + k * 16 + (c & 15)]
                          : W[(size_t)k * HIDD + c];
        Ws[c * LDS_STR + k] = (_Float16)w;
    }
    __syncthreads();

    const int wave = tid >> 5, lane = tid & 31;
    const int ct   = wave;            // column tile 0..7
    const int nrel = lane & 15;
    const int hi   = lane >> 4;
    const int col  = ct * 16 + nrel;

    // Hoist the 4 B fragments (reused across all 4 row tiles)
    v16h bfrag[4];
#pragma unroll
    for (int kc = 0; kc < 4; ++kc) {
        const _Float16* p = &Ws[col * LDS_STR + kc * 32 + hi * 8];
        bfrag[kc] = cat8(*(const v8h*)p, *(const v8h*)(p + 16));
    }
    if (MODE == 1) __syncthreads();   // Ws region is recycled as T below

    _Float16* T = Ws;                 // MODE 1 staging tile: T[c][r], stride T_STR

#pragma unroll 1
    for (int rt = 0; rt < 4; ++rt) {
        v8f acc = {0.f,0.f,0.f,0.f,0.f,0.f,0.f,0.f};
        const int arow = rt * 16 + nrel;
#pragma unroll
        for (int kc = 0; kc < 4; ++kc) {
            const _Float16* p = &As[arow * LDS_STR + kc * 32 + hi * 8];
            v16h a = cat8(*(const v8h*)p, *(const v8h*)(p + 16));
            acc = __builtin_amdgcn_wmma_f32_16x16x32_f16(
                      false, a, false, bfrag[kc], (short)0, acc, false, false);
        }
        if (MODE == 0) {
            const float bval = bias[col];
#pragma unroll
            for (int v = 0; v < 8; ++v) {
                const int mr  = hi ? (v + 8) : v;
                const int row = m0 + rt * 16 + mr;
                float val = acc[v] + bval;
                if (HAS_RES) val += residual[(size_t)row * HIDD + col];
                if (DO_GELU) val  = gelu_f(val);
                out32[(size_t)row * HIDD + col] = val;
            }
        } else {
            // acc[v] covers 8 consecutive output rows (r = rt*16 + hi*8 + v)
            v8h pk;
#pragma unroll
            for (int v = 0; v < 8; ++v) pk[v] = (_Float16)acc[v];
            *(v8h*)&T[col * T_STR + rt * 16 + hi * 8] = pk;
        }
    }

    if (MODE == 1) {
        __syncthreads();
        // Coalesced write-out: thread -> (col = tid>>1, half-row = tid&1)
        const int c  = tid >> 1;
        const int hs = (tid & 1) * 32;
        const int hh = c >> 4, vd = c & 15;
        const int b  = m0 >> 11, n0 = m0 & (NTOK - 1);
        _Float16* dst = out16 + ((size_t)(b * NH + hh) * VDIM + vd) * NTOK + n0 + hs;
        const _Float16* src = &T[c * T_STR + hs];
#pragma unroll
        for (int t = 0; t < 4; ++t)
            *(v8h*)(dst + t * 8) = *(const v8h*)(src + t * 8);
    }
}

// ---------------------------------------------------------------------------
// MHPA core: one wave per (b, h, 16-row tile). Streams m_dist in 32-wide
// chunks, builds exp(-m*r^2) (masked) directly in WMMA A-fragment layout,
// multiplies against transposed f16 V (B fragment), tracks row sums, then
// normalizes (single v_rcp) + fused GELU epilogue.
// ---------------------------------------------------------------------------
__global__ __launch_bounds__(256)
void attn_kernel(const float* __restrict__ m_dist, const _Float16* __restrict__ Vt,
                 const float* __restrict__ r_arr, const float* __restrict__ thr,
                 float* __restrict__ out) {
    const int wave = threadIdx.x >> 5, lane = threadIdx.x & 31;
    const int task = blockIdx.x * 8 + wave;     // 0..2047
    const int it = task & 127;
    const int h  = (task >> 7) & 7;
    const int b  = task >> 10;
    const int i0 = it * 16;

    const float r  = r_arr[h];
    const float r2 = r * r;
    const int nrel = lane & 15;
    const int hi   = lane >> 4;
    const int row  = i0 + nrel;                 // this lane's A-fragment row

    const float* mrow = m_dist + ((size_t)b * NTOK + row) * NTOK;
    const float  th   = thr ? thr[b * NTOK + row] : 3.4028235e38f;
    const _Float16* vrow = Vt + ((size_t)(b * NH + h) * VDIM + nrel) * NTOK;

    float rsum = 0.f;
    v8f acc = {0.f,0.f,0.f,0.f,0.f,0.f,0.f,0.f};

#pragma unroll 1
    for (int j0 = 0; j0 < NTOK; j0 += 32) {
        // speculative prefetch of next chunk (silently dropped past the end)
        __builtin_prefetch(mrow + j0 + 32, 0, 1);       // global_prefetch_b8

        const int ka = j0 + hi * 8;
        float mv[16];
        {
            v4f q0 = *(const v4f*)(mrow + ka);
            v4f q1 = *(const v4f*)(mrow + ka + 4);
            v4f q2 = *(const v4f*)(mrow + ka + 16);
            v4f q3 = *(const v4f*)(mrow + ka + 20);
#pragma unroll
            for (int t = 0; t < 4; ++t) {
                mv[t] = q0[t]; mv[4+t] = q1[t]; mv[8+t] = q2[t]; mv[12+t] = q3[t];
            }
        }
        v16h a;
#pragma unroll
        for (int t = 0; t < 16; ++t) {
            float e = (mv[t] <= th) ? __expf(-mv[t] * r2) : 0.0f;
            rsum += e;
            a[t] = (_Float16)e;
        }
        const int kb = j0 + hi * 8;
        v16h bf = cat8(*(const v8h*)(vrow + kb), *(const v8h*)(vrow + kb + 16));
        acc = __builtin_amdgcn_wmma_f32_16x16x32_f16(
                  false, a, false, bf, (short)0, acc, false, false);
    }

    // full row sum: lane L and lane L^16 cover complementary K subsets;
    // one v_rcp per lane, broadcast the inverse (not a div per output)
    const float rs  = rsum + swapx16_f(rsum);
    const float inv = __builtin_amdgcn_rcpf(rs);

#pragma unroll
    for (int v = 0; v < 8; ++v) {
        const int mr = hi ? (v + 8) : v;
        const float rinv = __shfl(inv, mr, 32);
        float o = acc[v] * rinv;
        o = gelu_f(o);                           // gelu fused into mhpa epilogue
        out[((size_t)(b * NTOK) + i0 + mr) * HIDD + h * VDIM + nrel] = o;
    }
}

// ---------------------------------------------------------------------------
// Final head: out(4096) = t2(4096x128) @ de_W2(128x1) + de_b2
// ---------------------------------------------------------------------------
__global__ __launch_bounds__(256)
void head_kernel(const float* __restrict__ t2, const float* __restrict__ W2,
                 const float* __restrict__ b2, float* __restrict__ out) {
    int r = blockIdx.x * 256 + threadIdx.x;
    if (r >= MROWS) return;
    float s = b2[0];
#pragma unroll 4
    for (int k = 0; k < HIDD; ++k) s += t2[(size_t)r * HIDD + k] * W2[k];
    out[r] = s;
}

// ---------------------------------------------------------------------------
extern "C" void kernel_launch(void* const* d_in, const int* in_sizes, int n_in,
                              void* d_out, int out_size, void* d_ws, size_t ws_size,
                              hipStream_t stream) {
    const float* m_dist  = (const float*)d_in[0];
    const float* inputs  = (const float*)d_in[1];
    const float* en_W    = (const float*)d_in[2];
    const float* en_b    = (const float*)d_in[3];
    const float* down_r  = (const float*)d_in[4];
    const float* down_w  = (const float*)d_in[5];
    const float* mlp1_W1 = (const float*)d_in[6];
    const float* mlp1_b1 = (const float*)d_in[7];
    const float* mlp1_W2 = (const float*)d_in[8];
    const float* mlp1_b2 = (const float*)d_in[9];
    const float* w1_W    = (const float*)d_in[10];
    const float* w1_b    = (const float*)d_in[11];
    const float* pa_r    = (const float*)d_in[12];
    const float* pa_w    = (const float*)d_in[13];
    const float* blk_W1  = (const float*)d_in[14];
    const float* blk_b1  = (const float*)d_in[15];
    const float* blk_W2  = (const float*)d_in[16];
    const float* blk_b2  = (const float*)d_in[17];
    const float* wi_W    = (const float*)d_in[18];
    const float* wi_b    = (const float*)d_in[19];
    const float* up_r    = (const float*)d_in[20];
    const float* up_w    = (const float*)d_in[21];
    const float* mlp2_W1 = (const float*)d_in[22];
    const float* mlp2_b1 = (const float*)d_in[23];
    const float* mlp2_W2 = (const float*)d_in[24];
    const float* mlp2_b2 = (const float*)d_in[25];
    const float* w2_W    = (const float*)d_in[26];
    const float* w2_b    = (const float*)d_in[27];
    const float* de_W1   = (const float*)d_in[28];
    const float* de_b1   = (const float*)d_in[29];
    const float* de_W2   = (const float*)d_in[30];
    const float* de_b2   = (const float*)d_in[31];

    // workspace carve-out
    size_t off = 0;
    auto wsa = [&](size_t bytes) -> void* {
        void* p = (char*)d_ws + off;
        off += (bytes + 255) & ~(size_t)255;
        return p;
    };
    float*     thr = (float*)wsa(MROWS * sizeof(float));
    float*     bEn = (float*)wsa((size_t)MROWS * HIDD * sizeof(float));
    float*     bH  = (float*)wsa((size_t)MROWS * HIDD * sizeof(float));
    float*     bR  = (float*)wsa((size_t)MROWS * HIDD * sizeof(float));
    float*     bT  = (float*)wsa((size_t)MROWS * HIDD * sizeof(float));
    float*     bX  = (float*)wsa((size_t)MROWS * HIDD * sizeof(float));
    _Float16*  Vt  = (_Float16*)wsa((size_t)BB * NH * VDIM * NTOK * sizeof(_Float16));
    (void)ws_size; (void)n_in; (void)in_sizes; (void)out_size;

    // percentile fraction: pos = 5/100 * (2048-1) = 102.35
    const double pos  = 5.0 / 100.0 * (double)(NTOK - 1);
    const float  frac = (float)(pos - 102.0);

    percentile_kernel<<<MROWS, 256, 0, stream>>>(m_dist, thr, frac);
    encoder_kernel<<<(MROWS * HIDD + 255) / 256, 256, 0, stream>>>(inputs, en_W, en_b, bEn);

    // One MHPA + MLP + residual-projection stage
    auto stage = [&](const float* x_in, const float* vw, const float* rr,
                     const float* thrp, const float* rW, const float* rb,
                     const float* W1, const float* b1, const float* W2,
                     const float* b2, float* x_out) {
        // V = x @ w (per head), stored transposed f16
        gemm128_t<1,1,0,0><<<64, 256, 0, stream>>>(x_in, vw, nullptr, nullptr,
                                                   nullptr, Vt);
        // h = gelu(softmax(-m*r^2 masked) @ V)
        attn_kernel<<<256, 256, 0, stream>>>(m_dist, Vt, rr, thrp, bH);
        // r = x @ rW + rb
        gemm128_t<0,0,0,0><<<64, 256, 0, stream>>>(x_in, rW, rb, nullptr,
                                                   bR, nullptr);
        // t = gelu(h @ W1 + b1)
        gemm128_t<0,0,0,1><<<64, 256, 0, stream>>>(bH, W1, b1, nullptr,
                                                   bT, nullptr);
        // x_out = gelu(t @ W2 + b2 + r)
        gemm128_t<0,0,1,1><<<64, 256, 0, stream>>>(bT, W2, b2, bR,
                                                   x_out, nullptr);
    };

    // down block (locality = 5 -> masked)
    stage(bEn, down_w, down_r, thr, w1_W, w1_b,
          mlp1_W1, mlp1_b1, mlp1_W2, mlp1_b2, bX);

    // NB = 4 interior blocks (locality = 200 -> unmasked)
    for (int i = 0; i < 4; ++i) {
        stage(bX, pa_w + (size_t)i * NH * HIDD * VDIM, pa_r + (size_t)i * NH,
              nullptr, wi_W + (size_t)i * HIDD * HIDD, wi_b + (size_t)i * HIDD,
              blk_W1 + (size_t)i * HIDD * HIDD, blk_b1 + (size_t)i * HIDD,
              blk_W2 + (size_t)i * HIDD * HIDD, blk_b2 + (size_t)i * HIDD, bX);
    }

    // up block (locality = 5 -> masked); reuse bEn as "de"
    stage(bX, up_w, up_r, thr, w2_W, w2_b,
          mlp2_W1, mlp2_b1, mlp2_W2, mlp2_b2, bEn);

    // decoder MLP: t2 = gelu(de @ de_W1 + de_b1); out = t2 @ de_W2 + de_b2
    gemm128_t<0,0,0,1><<<64, 256, 0, stream>>>(bEn, de_W1, de_b1, nullptr,
                                               bT, nullptr);
    head_kernel<<<(MROWS + 255) / 256, 256, 0, stream>>>(bT, de_W2, de_b2,
                                                         (float*)d_out);
}
